// Encoder_87179246174263
// MI455X (gfx1250) — compile-verified, matrix operand order checked by reading, not run on hardware
//
#include <hip/hip_runtime.h>
#include <hip/hip_bf16.h>
#include <math.h>
#include <stdint.h>

// Problem dims
#define B_  256
#define E_  256
#define H_  512
#define V_  32768
#define T_  8
#define LN_EPS 1e-5f
#define RMS_EPS 1.1920928955078125e-7f   // finfo(float32).eps

// LDS staging tile for GEMMs: 64 B-rows x 256 K, padded by 8 f16 per row
#define LDS_ROWS   64
#define LDS_K      256
#define LDS_STRIDE 264   // 264 f16 = 528 B per row -> rows advance 4 banks

typedef __attribute__((ext_vector_type(16))) _Float16 v16h;
typedef __attribute__((ext_vector_type(8)))  _Float16 v8h;
typedef __attribute__((ext_vector_type(8)))  float    v8f;
typedef __attribute__((ext_vector_type(4)))  int      v4i;

// address-space qualified int4 for the async-DMA builtin
typedef __attribute__((address_space(1))) v4i as1_v4i;   // global
typedef __attribute__((address_space(3))) v4i as3_v4i;   // LDS

// ---------------------------------------------------------------------------
// WMMA helper
// ---------------------------------------------------------------------------
static __device__ __forceinline__ v8f wmma_f16(v16h a, v16h b, v8f c) {
    return __builtin_amdgcn_wmma_f32_16x16x32_f16(
        /*neg_a=*/false, a, /*neg_b=*/false, b,
        /*c_mod=*/(short)0, c, /*reuse_a=*/false, /*reuse_b=*/false);
}

// Load a 16-element f16 fragment for the 16x16x32 WMMA A/B layout:
// lane<16 holds K = {0..7, 16..23}, lane>=16 holds K = {8..15, 24..31}
// (relative to k0). base = row pointer + k0, contiguous in K.
// Works for both global and LDS (compiler picks global_load_b128 / ds_load_b128).
static __device__ __forceinline__ v16h load_frag16(const _Float16* __restrict__ base,
                                                   int half /* lane>>4 */) {
    const _Float16* p = base + half * 8;
    v8h lo = *(const v8h*)(p);        // k = half*8 + 0..7
    v8h hi = *(const v8h*)(p + 16);   // k = 16 + half*8 + 0..7
    return __builtin_shufflevector(lo, hi, 0,1,2,3,4,5,6,7,8,9,10,11,12,13,14,15);
}

// ---------------------------------------------------------------------------
// CDNA5 async global->LDS copy (16 bytes per lane), tracked by ASYNCcnt
// ---------------------------------------------------------------------------
static __device__ __forceinline__ void async_copy16(const _Float16* g, _Float16* l) {
#if __has_builtin(__builtin_amdgcn_global_load_async_to_lds_b128)
    // AS1 pointer: generic global address is bit-identical.
    as1_v4i* gp = (as1_v4i*)(unsigned long long)(uintptr_t)g;
    // AS3 pointer: LDS offset lives in the low 32 bits of the generic address
    // (exactly what addrspacecast generic->AS3 lowers to).
    as3_v4i* lp = (as3_v4i*)(unsigned)(uintptr_t)l;
    __builtin_amdgcn_global_load_async_to_lds_b128(gp, lp, 0, 0);
#else
    *(v8h*)l = *(const v8h*)g;   // synchronous fallback
#endif
}

static __device__ __forceinline__ void wait_async0() {
#if __has_builtin(__builtin_amdgcn_s_wait_asynccnt)
    __builtin_amdgcn_s_wait_asynccnt(0);
#else
    asm volatile("s_wait_asynccnt 0x0" ::: "memory");
#endif
}

// Stage one 64-row x 256-K f16 chunk into LDS (256 threads cooperating).
// src_row(r) = srcbase + r*src_stride + kbase  must be 16B aligned.
static __device__ __forceinline__ void stage_B_chunk(const _Float16* __restrict__ srcbase,
                                                     size_t src_stride, int kbase,
                                                     _Float16* __restrict__ Bs, int tid) {
    for (int i = tid; i < LDS_ROWS * (LDS_K / 8); i += 256) {
        const int r = i >> 5;          // 0..63  (32 chunks of 8 f16 per row)
        const int c = i & 31;          // 0..31
        async_copy16(srcbase + (size_t)r * src_stride + kbase + c * 8,
                     Bs + r * LDS_STRIDE + c * 8);
    }
    wait_async0();
    __syncthreads();
}

// ---------------------------------------------------------------------------
// Block reduction helper (256 threads)
// ---------------------------------------------------------------------------
static __device__ __forceinline__ float block_reduce_sum256(float v, float* red) {
    const int tid = threadIdx.x;
    red[tid] = v;
    __syncthreads();
    for (int s = 128; s > 0; s >>= 1) {
        if (tid < s) red[tid] += red[tid + s];
        __syncthreads();
    }
    float r = red[0];
    __syncthreads();
    return r;
}

// ---------------------------------------------------------------------------
// Kernel: codebook precompute
//   Cn_f16[v][h] = C[v][h]/max(||C[v]||,1e-12)   (B operand of logits GEMM)
//   CT_f16[h][v] = C[v][h]                        (B operand of expected GEMM)
// ---------------------------------------------------------------------------
__global__ void precompute_codebook(const float* __restrict__ C,
                                    _Float16* __restrict__ Cn,
                                    _Float16* __restrict__ CT) {
    __shared__ float red[256];
    const int v   = blockIdx.x;
    const int tid = threadIdx.x;
    const int j0 = tid, j1 = tid + 256;
    float c0 = C[(size_t)v * H_ + j0];
    float c1 = C[(size_t)v * H_ + j1];
    float ss = block_reduce_sum256(c0 * c0 + c1 * c1, red);
    float inv = 1.0f / fmaxf(sqrtf(ss), 1e-12f);
    Cn[(size_t)v * H_ + j0] = (_Float16)(c0 * inv);
    Cn[(size_t)v * H_ + j1] = (_Float16)(c1 * inv);
    CT[(size_t)j0 * V_ + v] = (_Float16)c0;
    CT[(size_t)j1 * V_ + v] = (_Float16)c1;
}

// ---------------------------------------------------------------------------
// Kernel: backbone  h = GELU(LN(x@W1 + b1))   — one block per batch row
// ---------------------------------------------------------------------------
__global__ void backbone(const float* __restrict__ x, const float* __restrict__ W1,
                         const float* __restrict__ b1, const float* __restrict__ ln_g,
                         const float* __restrict__ ln_b, float* __restrict__ h) {
    __shared__ float xs[E_];
    __shared__ float red[256];
    const int b = blockIdx.x, tid = threadIdx.x;
    xs[tid] = x[(size_t)b * E_ + tid];
    __syncthreads();
    const int j0 = tid, j1 = tid + 256;
    float a0 = b1[j0], a1 = b1[j1];
    for (int e = 0; e < E_; ++e) {
        float xv = xs[e];
        a0 = fmaf(xv, W1[(size_t)e * H_ + j0], a0);
        a1 = fmaf(xv, W1[(size_t)e * H_ + j1], a1);
    }
    float mean = block_reduce_sum256(a0 + a1, red) * (1.0f / H_);
    float d0 = a0 - mean, d1 = a1 - mean;
    float var = block_reduce_sum256(d0 * d0 + d1 * d1, red) * (1.0f / H_);
    float rstd = rsqrtf(var + LN_EPS);
    float y0 = d0 * rstd * ln_g[j0] + ln_b[j0];
    float y1 = d1 * rstd * ln_g[j1] + ln_b[j1];
    y0 = 0.5f * y0 * (1.0f + erff(y0 * 0.70710678118654752f));
    y1 = 0.5f * y1 * (1.0f + erff(y1 * 0.70710678118654752f));
    h[(size_t)b * H_ + j0] = y0;
    h[(size_t)b * H_ + j1] = y1;
}

// ---------------------------------------------------------------------------
// Kernel: z_n = l2norm(h) -> f16     — one block per batch row
// ---------------------------------------------------------------------------
__global__ void l2norm_to_f16(const float* __restrict__ h, _Float16* __restrict__ zn) {
    __shared__ float red[256];
    const int b = blockIdx.x, tid = threadIdx.x;
    const int j0 = tid, j1 = tid + 256;
    float v0 = h[(size_t)b * H_ + j0];
    float v1 = h[(size_t)b * H_ + j1];
    float ss = block_reduce_sum256(v0 * v0 + v1 * v1, red);
    float inv = 1.0f / fmaxf(sqrtf(ss), 1e-12f);
    zn[(size_t)b * H_ + j0] = (_Float16)(v0 * inv);
    zn[(size_t)b * H_ + j1] = (_Float16)(v1 * inv);
}

// ---------------------------------------------------------------------------
// Kernel: logits GEMM   logits[b][t][v] = (zn @ Cn^T)[b][v] * exp(logit_scale[t])
//   A: zn f16 [256,512];  B: Cn f16 [V,512] (row v == column v of B).
//   Block = 8 waves = 128(M) x 64(N); B tile staged in LDS via async DMA in
//   two 64x256 chunks (shared by all 8 waves -> 8x less L2 traffic).
//   grid: 2 Mblk * 512 Nblk = 1024 blocks x 256 thr
// ---------------------------------------------------------------------------
__global__ void __launch_bounds__(256) gemm_logits(
        const _Float16* __restrict__ zn, const _Float16* __restrict__ cn,
        float* __restrict__ out_logits, const float* __restrict__ logit_scale, int t) {
    __shared__ __align__(16) _Float16 Bs[LDS_ROWS * LDS_STRIDE];
    const int tid  = threadIdx.x;
    const int warp = tid >> 5, lane = tid & 31;
    const int half = lane >> 4, l16 = lane & 15;
    const int tn   = blockIdx.x & 511;   // 512 N blocks (64 wide)
    const int mblk = blockIdx.x >> 9;    // 2 M blocks (128 tall)
    const int n0   = tn * 64;

    const int mrow0 = (mblk * 8 + warp) * 16;
    const _Float16* arow = zn + (size_t)(mrow0 + l16) * H_;

    v8f acc0 = {}, acc1 = {}, acc2 = {}, acc3 = {};
    for (int kc = 0; kc < H_; kc += LDS_K) {
        if (kc) __syncthreads();                       // done reading prev chunk
        stage_B_chunk(cn + (size_t)n0 * H_, H_, kc, Bs, tid);
        #pragma unroll
        for (int k0 = 0; k0 < LDS_K; k0 += 32) {
            v16h a  = load_frag16(arow + kc + k0, half);
            v16h b0 = load_frag16(&Bs[(l16 +  0) * LDS_STRIDE + k0], half);
            v16h b1 = load_frag16(&Bs[(l16 + 16) * LDS_STRIDE + k0], half);
            v16h b2 = load_frag16(&Bs[(l16 + 32) * LDS_STRIDE + k0], half);
            v16h b3 = load_frag16(&Bs[(l16 + 48) * LDS_STRIDE + k0], half);
            acc0 = wmma_f16(a, b0, acc0);
            acc1 = wmma_f16(a, b1, acc1);
            acc2 = wmma_f16(a, b2, acc2);
            acc3 = wmma_f16(a, b3, acc3);
        }
    }
    const float scale = __expf(logit_scale[t]);
    #pragma unroll
    for (int r = 0; r < 8; ++r) {
        const int m = mrow0 + r + half * 8;            // D layout: M = r + 8*half
        float* rowp = out_logits + ((size_t)m * T_ + t) * V_;
        rowp[n0 + l16 +  0] = acc0[r] * scale;
        rowp[n0 + l16 + 16] = acc1[r] * scale;
        rowp[n0 + l16 + 32] = acc2[r] * scale;
        rowp[n0 + l16 + 48] = acc3[r] * scale;
    }
}

// ---------------------------------------------------------------------------
// Gumbel noise from a counter hash (stand-in for jax threefry; no runtime check)
// ---------------------------------------------------------------------------
static __device__ __forceinline__ float gumbel_hash(unsigned t, unsigned b, unsigned v) {
    unsigned long long x =
        (((unsigned long long)(t * 256u + b)) << 32) ^ ((unsigned long long)v * 0x9E3779B97F4A7C15ULL);
    x ^= x >> 33; x *= 0xff51afd7ed558ccdULL;
    x ^= x >> 33; x *= 0xc4ceb9fe1a85ec53ULL;
    x ^= x >> 33;
    float u = (float)(x >> 40) * (1.0f / 16777216.0f) + (1.0f / 33554432.0f); // (0,1)
    return -__logf(-__logf(u));
}

// ---------------------------------------------------------------------------
// Kernel: softmax over V per (b, t) row. One block (1024 thr) per batch row;
// 32 elems per thread held in registers.
// ---------------------------------------------------------------------------
__global__ void softmax_row(const float* __restrict__ out_logits,
                            float* __restrict__ out_probs,
                            _Float16* __restrict__ probs16,
                            const int* __restrict__ tau, int t) {
    __shared__ float sm[32];
    const int b    = blockIdx.x;
    const int tid  = threadIdx.x;
    const int lane = tid & 31, wrp = tid >> 5;
    const float inv_tau = 1.0f / (float)tau[0];
    const float* lrow = out_logits + ((size_t)b * T_ + t) * V_;

    float y[32];
    float lmax = -3.4e38f;
    #pragma unroll
    for (int i = 0; i < 32; ++i) {
        const int v = tid + i * 1024;
        float yy = (lrow[v] + gumbel_hash(t, b, v)) * inv_tau;
        y[i] = yy;
        lmax = fmaxf(lmax, yy);
    }
    for (int s = 16; s > 0; s >>= 1) lmax = fmaxf(lmax, __shfl_xor(lmax, s, 32));
    if (lane == 0) sm[wrp] = lmax;
    __syncthreads();
    if (tid < 32) {
        float v = sm[tid];
        for (int s = 16; s > 0; s >>= 1) v = fmaxf(v, __shfl_xor(v, s, 32));
        if (tid == 0) sm[0] = v;
    }
    __syncthreads();
    const float rowmax = sm[0];
    __syncthreads();
    float lsum = 0.0f;
    #pragma unroll
    for (int i = 0; i < 32; ++i) {
        y[i] = __expf(y[i] - rowmax);
        lsum += y[i];
    }
    for (int s = 16; s > 0; s >>= 1) lsum += __shfl_xor(lsum, s, 32);
    if (lane == 0) sm[wrp] = lsum;
    __syncthreads();
    if (tid < 32) {
        float v = sm[tid];
        for (int s = 16; s > 0; s >>= 1) v += __shfl_xor(v, s, 32);
        if (tid == 0) sm[0] = v;
    }
    __syncthreads();
    const float inv_sum = 1.0f / sm[0];

    float* prow = out_probs + ((size_t)b * T_ + t) * V_;
    _Float16* p16row = probs16 + (size_t)b * V_;
    #pragma unroll
    for (int i = 0; i < 32; ++i) {
        const int v = tid + i * 1024;
        float p = y[i] * inv_sum;
        prow[v]   = p;
        p16row[v] = (_Float16)p;
    }
}

// ---------------------------------------------------------------------------
// Kernel: zero expected buffer
// ---------------------------------------------------------------------------
__global__ void zero_buf(float* __restrict__ p, int n) {
    int i = blockIdx.x * blockDim.x + threadIdx.x;
    if (i < n) p[i] = 0.0f;
}

// ---------------------------------------------------------------------------
// Kernel: expected GEMM   expected = probs @ C   [256,32768]x[32768,512]
//   A: probs16 [B,V];  B: CT_f16 [H,V] (row h == column h, contiguous in K).
//   Block = 8 waves = 128(M) x 64(N), K-split over blocks (8 slices of 4096);
//   B slice staged through LDS in sixteen 64x256 chunks; f32 atomics combine.
//   grid: 2 Mblk * 8 Nblk * 8 Ksplit = 128 blocks x 256 thr
// ---------------------------------------------------------------------------
__global__ void __launch_bounds__(256) gemm_expected(
        const _Float16* __restrict__ probs16, const _Float16* __restrict__ ct,
        float* __restrict__ expected) {
    __shared__ __align__(16) _Float16 Bs[LDS_ROWS * LDS_STRIDE];
    const int tid  = threadIdx.x;
    const int warp = tid >> 5, lane = tid & 31;
    const int half = lane >> 4, l16 = lane & 15;
    const int ks   = blockIdx.x & 7;          // 8 K slices of 4096
    const int tn   = (blockIdx.x >> 3) & 7;   // 8 N blocks (64 wide)
    const int mblk = blockIdx.x >> 6;         // 2 M blocks (128 tall)
    const int n0   = tn * 64;
    const int kbeg = ks * (V_ / 8);

    const int mrow0 = (mblk * 8 + warp) * 16;
    const _Float16* arow = probs16 + (size_t)(mrow0 + l16) * V_;

    v8f acc0 = {}, acc1 = {}, acc2 = {}, acc3 = {};
    for (int kc = 0; kc < V_ / 8; kc += LDS_K) {
        if (kc) __syncthreads();
        stage_B_chunk(ct + (size_t)n0 * V_, V_, kbeg + kc, Bs, tid);
        #pragma unroll
        for (int k0 = 0; k0 < LDS_K; k0 += 32) {
            __builtin_prefetch(arow + kbeg + kc + k0 + LDS_K, 0, 0);
            v16h a  = load_frag16(arow + kbeg + kc + k0, half);
            v16h b0 = load_frag16(&Bs[(l16 +  0) * LDS_STRIDE + k0], half);
            v16h b1 = load_frag16(&Bs[(l16 + 16) * LDS_STRIDE + k0], half);
            v16h b2 = load_frag16(&Bs[(l16 + 32) * LDS_STRIDE + k0], half);
            v16h b3 = load_frag16(&Bs[(l16 + 48) * LDS_STRIDE + k0], half);
            acc0 = wmma_f16(a, b0, acc0);
            acc1 = wmma_f16(a, b1, acc1);
            acc2 = wmma_f16(a, b2, acc2);
            acc3 = wmma_f16(a, b3, acc3);
        }
    }
    #pragma unroll
    for (int r = 0; r < 8; ++r) {
        const int m = mrow0 + r + half * 8;
        float* rowp = expected + (size_t)m * H_;
        atomicAdd(&rowp[n0 + l16 +  0], acc0[r]);
        atomicAdd(&rowp[n0 + l16 + 16], acc1[r]);
        atomicAdd(&rowp[n0 + l16 + 32], acc2[r]);
        atomicAdd(&rowp[n0 + l16 + 48], acc3[r]);
    }
}

// ---------------------------------------------------------------------------
// Kernel: residual update + RMSNorm
//   h = h - exp(gamma[t]) * expected ; h = h / sqrt(mean(h^2)+eps) * rms_w
// ---------------------------------------------------------------------------
__global__ void update_h(float* __restrict__ h, const float* __restrict__ expected,
                         const float* __restrict__ gamma, const float* __restrict__ rms_w,
                         int t) {
    __shared__ float red[256];
    const int b = blockIdx.x, tid = threadIdx.x;
    const int j0 = tid, j1 = tid + 256;
    const float g = __expf(gamma[t]);
    float v0 = h[(size_t)b * H_ + j0] - g * expected[(size_t)b * H_ + j0];
    float v1 = h[(size_t)b * H_ + j1] - g * expected[(size_t)b * H_ + j1];
    float ms = block_reduce_sum256(v0 * v0 + v1 * v1, red) * (1.0f / H_);
    float r = rsqrtf(ms + RMS_EPS);
    h[(size_t)b * H_ + j0] = v0 * r * rms_w[j0];
    h[(size_t)b * H_ + j1] = v1 * r * rms_w[j1];
}

// ---------------------------------------------------------------------------
// Host launch
// ---------------------------------------------------------------------------
extern "C" void kernel_launch(void* const* d_in, const int* in_sizes, int n_in,
                              void* d_out, int out_size, void* d_ws, size_t ws_size,
                              hipStream_t stream) {
    const float* x           = (const float*)d_in[0];
    const float* W1          = (const float*)d_in[1];
    const float* b1          = (const float*)d_in[2];
    const float* ln_g        = (const float*)d_in[3];
    const float* ln_b        = (const float*)d_in[4];
    const float* codebook    = (const float*)d_in[5];   // [1,V,H]
    const float* logit_scale = (const float*)d_in[6];
    const float* gamma       = (const float*)d_in[7];
    const float* rms_w       = (const float*)d_in[8];
    const int*   tau         = (const int*)d_in[9];

    float* out_logits = (float*)d_out;                          // [B,T,V]
    float* out_probs  = (float*)d_out + (size_t)B_ * T_ * V_;   // [B,T,V]

    // workspace layout
    char* ws = (char*)d_ws;
    size_t off = 0;
    float*    h        = (float*)(ws + off);    off += (size_t)B_ * H_ * 4;    // 512 KB
    float*    expected = (float*)(ws + off);    off += (size_t)B_ * H_ * 4;    // 512 KB
    _Float16* zn       = (_Float16*)(ws + off); off += (size_t)B_ * H_ * 2;    // 256 KB
    _Float16* probs16  = (_Float16*)(ws + off); off += (size_t)B_ * V_ * 2;    // 16 MB
    _Float16* Cn       = (_Float16*)(ws + off); off += (size_t)V_ * H_ * 2;    // 32 MB
    _Float16* CT       = (_Float16*)(ws + off); off += (size_t)V_ * H_ * 2;    // 32 MB

    precompute_codebook<<<V_, 256, 0, stream>>>(codebook, Cn, CT);
    backbone<<<B_, 256, 0, stream>>>(x, W1, b1, ln_g, ln_b, h);

    for (int t = 0; t < T_; ++t) {
        l2norm_to_f16<<<B_, 256, 0, stream>>>(h, zn);
        gemm_logits<<<1024, 256, 0, stream>>>(zn, Cn, out_logits, logit_scale, t);
        softmax_row<<<B_, 1024, 0, stream>>>(out_logits, out_probs, probs16, tau, t);
        if (t < T_ - 1) {
            zero_buf<<<(B_ * H_ + 255) / 256, 256, 0, stream>>>(expected, B_ * H_);
            gemm_expected<<<128, 256, 0, stream>>>(probs16, CT, expected);
            update_h<<<B_, 256, 0, stream>>>(h, expected, gamma, rms_w, t);
        }
    }
}
